// flash_attention_53549652247101
// MI455X (gfx1250) — matrix operand order repaired and checked by reading, not compiled, not run on hardware
//
#include <hip/hip_runtime.h>
#include <stdint.h>

typedef __attribute__((ext_vector_type(16))) _Float16 v16h;
typedef __attribute__((ext_vector_type(8)))  _Float16 v8h;
typedef __attribute__((ext_vector_type(4)))  _Float16 v4h;
typedef __attribute__((ext_vector_type(8)))  float    v8f;
typedef __attribute__((ext_vector_type(4)))  float    v4f;

#define B_    2
#define H_    16
#define L_    2048
#define D_    64
#define BQ    128
#define BKV   128
#define NKV   (L_ / BKV)   // 16
#define KSTR  72           // Ks row stride (halfs), 64 + 8 pad
#define VSTR  136          // Vt row stride (halfs), 128 + 8 pad
#define PSTR  72           // P staging row stride (halfs), 64-col half-tile + pad

__device__ __forceinline__ uint32_t hash32(uint32_t x) {
  x ^= x >> 16; x *= 0x7feb352dU;
  x ^= x >> 15; x *= 0x846ca68bU;
  x ^= x >> 16;
  return x;
}

// B-layout fragment (16x16x32 f16): chunk K[8*hi .. +8) -> v0..3, K[16+8*hi .. +8) -> v4..7.
__device__ __forceinline__ v16h load_b_frag(const _Float16* p) {
  v8h lo = *(const v8h*)p;
  v8h hi = *(const v8h*)(p + 16);
  v16h r;
#pragma unroll
  for (int i = 0; i < 8; ++i) { r[i] = lo[i]; r[i + 8] = hi[i]; }
  return r;
}

// A-layout fragment: chunk K[8*hi .. +8) -> v0..3, K[16+8*hi .. +8) -> v4..7.
// p must already point at row base + 8*hi.
__device__ __forceinline__ v16h load_a_frag(const _Float16* p) {
  v8h lo = *(const v8h*)p;
  v8h hi = *(const v8h*)(p + 16);
  v16h r;
#pragma unroll
  for (int i = 0; i < 8; ++i) { r[i] = lo[i]; r[i + 8] = hi[i]; }
  return r;
}

__global__ void __launch_bounds__(256)
fa_fwd_kernel(const float* __restrict__ Q, const float* __restrict__ Kp,
              const float* __restrict__ Vp, float* __restrict__ Out,
              float* __restrict__ Lout, float* __restrict__ Mout)
{
  __shared__ _Float16 Ks[BKV * KSTR];     // K block, row-major f16
  __shared__ _Float16 Vt[D_ * VSTR];      // V block, transposed f16 (d-major)
  __shared__ _Float16 Ps[8 * 16 * PSTR];  // per-wave P staging (64-col halves)

  const int tid    = threadIdx.x;
  const int wv     = tid >> 5;
  const int ln     = tid & 31;
  const int lanelo = ln & 15;
  const int hi     = ln >> 4;
  const int bh     = blockIdx.y;   // b*H + h, 0..31
  const int qb     = blockIdx.x;   // Q block, 0..15
  const int r0     = wv * 16;      // wave's row strip within Q block

  const long baseQ  = ((long)bh * L_ + (long)qb * BQ) * D_;
  const long baseKV = (long)bh * L_ * D_;

  // ---- Q A-fragments for this wave's 16 rows (fp32 -> f16), reused all blocks
  v16h aq[2];
  {
    const float* qrow = Q + baseQ + (long)(r0 + lanelo) * D_;
#pragma unroll
    for (int kd = 0; kd < 2; ++kd) {
      const float* p0 = qrow + kd * 32 + 8 * hi;
      v4f f0 = *(const v4f*)(p0);
      v4f f1 = *(const v4f*)(p0 + 4);
      v4f f2 = *(const v4f*)(p0 + 16);
      v4f f3 = *(const v4f*)(p0 + 20);
      v16h a;
#pragma unroll
      for (int i = 0; i < 4; ++i) {
        a[i]      = (_Float16)f0[i];
        a[i + 4]  = (_Float16)f1[i];
        a[i + 8]  = (_Float16)f2[i];
        a[i + 12] = (_Float16)f3[i];
      }
      aq[kd] = a;
    }
  }

  float m_run[8], l_run[8];
  v8f oacc[4] = {};
#pragma unroll
  for (int v = 0; v < 8; ++v) { m_run[v] = -1e30f; l_run[v] = 0.0f; }

  _Float16* psw = Ps + wv * 16 * PSTR;

  for (int c = 0; c < NKV; ++c) {
    const float* kblk = Kp + baseKV + (long)c * BKV * D_;
    const float* vblk = Vp + baseKV + (long)c * BKV * D_;

    // ---- cooperative fill: Ks row-major, Vt transposed (f32 -> f16) ----
#pragma unroll
    for (int it = 0; it < 8; ++it) {
      int g   = tid + it * 256;      // float4 index in 128x64 tile (2048 total)
      int row = g >> 4;
      int col = (g & 15) * 4;
      v4f kf = *(const v4f*)(kblk + row * D_ + col);
      v4f vf = *(const v4f*)(vblk + row * D_ + col);
      v4h kh;
#pragma unroll
      for (int i = 0; i < 4; ++i) kh[i] = (_Float16)kf[i];
      *(v4h*)(Ks + row * KSTR + col) = kh;
#pragma unroll
      for (int i = 0; i < 4; ++i)
        Vt[(col + i) * VSTR + row] = (_Float16)vf[i];
    }
    __syncthreads();

    // prefetch next K/V block (128B cacheline per thread covers each 32KB tile)
    if (c + 1 < NKV) {
      __builtin_prefetch(kblk + BKV * D_ + tid * 32, 0, 3);
      __builtin_prefetch(vblk + BKV * D_ + tid * 32, 0, 3);
    }

    // ---- S = Q K^T : 16x128 strip per wave, fp32 accum ----
    v8f sacc[8] = {};
#pragma unroll
    for (int kd = 0; kd < 2; ++kd) {
#pragma unroll
      for (int t = 0; t < 8; ++t) {
        v16h bk = load_b_frag(Ks + (t * 16 + lanelo) * KSTR + kd * 32 + 8 * hi);
        sacc[t] = __builtin_amdgcn_wmma_f32_16x16x32_f16(
            false, aq[kd], false, bk, (short)0, sacc[t], false, false);
      }
    }

    // ---- threshold mask, per-row max over 128 cols ----
    float mc[8], lc[8];
#pragma unroll
    for (int v = 0; v < 8; ++v) {
      float mx = -1e30f;
#pragma unroll
      for (int t = 0; t < 8; ++t) {
        float s = sacc[t][v];
        s = (s >= 0.5f) ? 0.0f : s;
        sacc[t][v] = s;
        mx = fmaxf(mx, s);
      }
#pragma unroll
      for (int m = 1; m <= 8; m <<= 1) mx = fmaxf(mx, __shfl_xor(mx, m, 32));
      mc[v] = mx;
    }

    // ---- P = exp(S - m), row sums (+eps) ----
#pragma unroll
    for (int v = 0; v < 8; ++v) {
      float sum = 0.0f;
#pragma unroll
      for (int t = 0; t < 8; ++t) {
        float p = __expf(sacc[t][v] - mc[v]);
        sacc[t][v] = p;
        sum += p;
      }
#pragma unroll
      for (int m = 1; m <= 8; m <<= 1) sum += __shfl_xor(sum, m, 32);
      lc[v] = sum + 1e-10f;
    }

    // ---- PV = P @ V, two 64-column halves through LDS re-layout ----
    v8f pvc[4] = {};
#pragma unroll
    for (int half = 0; half < 2; ++half) {
      // stage P cols [half*64, +64) in C-layout -> row-major LDS
#pragma unroll
      for (int v = 0; v < 8; ++v) {
#pragma unroll
        for (int t = 0; t < 4; ++t) {
          psw[(v + 8 * hi) * PSTR + t * 16 + lanelo] =
              (_Float16)sacc[half * 4 + t][v];
        }
      }
      asm volatile("s_wait_dscnt 0x0" ::: "memory");
#pragma unroll
      for (int jb = 0; jb < 2; ++jb) {
        v16h ap = load_a_frag(psw + lanelo * PSTR + jb * 32 + 8 * hi);
#pragma unroll
        for (int t = 0; t < 4; ++t) {
          v16h bv = load_b_frag(Vt + (t * 16 + lanelo) * VSTR
                                + half * 64 + jb * 32 + 8 * hi);
          pvc[t] = __builtin_amdgcn_wmma_f32_16x16x32_f16(
              false, ap, false, bv, (short)0, pvc[t], false, false);
        }
      }
      asm volatile("s_wait_dscnt 0x0" ::: "memory");
    }

    // ---- dropout on per-block PV + online rescale/accumulate ----
#pragma unroll
    for (int v = 0; v < 8; ++v) {
      float mn    = fmaxf(m_run[v], mc[v]);
      float alpha = __expf(m_run[v] - mn);
      float beta  = __expf(mc[v] - mn);
      l_run[v] = alpha * l_run[v] + beta * lc[v];
      m_run[v] = mn;
      uint32_t qrow = (uint32_t)(r0 + v + 8 * hi);
      uint32_t base = ((((uint32_t)bh * 16u + (uint32_t)qb) * 128u + qrow) * 16u
                       + (uint32_t)c) * 64u;
#pragma unroll
      for (int t = 0; t < 4; ++t) {
        float pv = pvc[t][v];
        uint32_t h = hash32((base + (uint32_t)(t * 16 + lanelo)) * 0x9E3779B9u
                            + 0x2A2Au);
        pv = ((h >> 8) < 13421773u) ? pv * 1.25f : 0.0f;  // keep p=0.8, /0.8
        oacc[t][v] = alpha * oacc[t][v] + beta * pv;
      }
    }
    __syncthreads();   // protect Ks/Vt before next block's fill
  }

  // ---- epilogue: O = acc / l, plus l and m ----
  const long rowbase = (long)bh * L_ + (long)qb * BQ + r0;
#pragma unroll
  for (int v = 0; v < 8; ++v) {
    long grow = rowbase + v + 8 * hi;
    float inv = 1.0f / l_run[v];
#pragma unroll
    for (int t = 0; t < 4; ++t)
      Out[grow * D_ + t * 16 + lanelo] = oacc[t][v] * inv;
    if (lanelo == 0) {
      Lout[grow] = l_run[v];
      Mout[grow] = m_run[v];
    }
  }
}

extern "C" void kernel_launch(void* const* d_in, const int* in_sizes, int n_in,
                              void* d_out, int out_size, void* d_ws, size_t ws_size,
                              hipStream_t stream) {
  (void)in_sizes; (void)n_in; (void)out_size; (void)d_ws; (void)ws_size;
  const float* Q = (const float*)d_in[0];
  const float* K = (const float*)d_in[1];
  const float* V = (const float*)d_in[2];
  float* O = (float*)d_out;
  float* L = O + (size_t)B_ * H_ * L_ * D_;
  float* M = L + (size_t)B_ * H_ * L_;
  dim3 grid(L_ / BQ, B_ * H_);
  dim3 block(256);
  hipLaunchKernelGGL(fa_fwd_kernel, grid, block, 0, stream, Q, K, V, O, L, M);
}
